// inner_GNN_87067577024886
// MI455X (gfx1250) — matrix operand order repaired and checked by reading, not compiled, not run on hardware
//
#include <hip/hip_runtime.h>
#include <hip/hip_bf16.h>
#include <math.h>

#define N_NODES 200000
#define N_EDGES 6400000
#define EP_EDGES (N_EDGES + N_NODES)   // with self loops
#define F_IN 256
#define NEG_SLOPE 0.2f
#define BN_EPS 1e-5f
#define ROW_PAD 260                    // 256 + 4 DW TDM pad -> bank-conflict-free LDS rows
#define TILES (N_NODES / 16)           // 12500, exact
#define WAVES_PER_BLK 2

typedef __attribute__((ext_vector_type(16))) _Float16 v16h;
typedef __attribute__((ext_vector_type(8)))  float    v8f;
typedef __attribute__((ext_vector_type(4)))  unsigned v4u;
typedef __attribute__((ext_vector_type(8)))  int      v8i;
typedef __attribute__((ext_vector_type(4)))  int      v4i;

// Order-preserving float->uint encoding so segment-max can use atomicMax(u32).
__device__ __forceinline__ unsigned enc_f32(float f) {
  unsigned u = __float_as_uint(f);
  return (u & 0x80000000u) ? ~u : (u | 0x80000000u);
}
__device__ __forceinline__ float dec_f32(unsigned e) {
  unsigned u = (e & 0x80000000u) ? (e & 0x7fffffffu) : ~e;
  return __uint_as_float(u);
}

// ---------------- init (graph-replay safe: re-zero accumulators every call) --
__global__ void k_init(unsigned* smax1, float* den1, float* h1,
                       unsigned* smax2, float* den2, float* h2, float* stats) {
  int i = blockIdx.x * blockDim.x + threadIdx.x;
  if (i < N_NODES) {
    smax1[i] = 0u; den1[i] = 0.f; h1[i] = 0.f;
    smax2[i] = 0u; den2[i] = 0.f; h2[i] = 0.f;
  }
  if (i < 4) stats[i] = 0.f;
}

// ---------------- int64 edge_index -> int32 src/dst with self loops ---------
__global__ void k_edges(const long long* __restrict__ ei,
                        int* __restrict__ src, int* __restrict__ dst) {
  int e = blockIdx.x * blockDim.x + threadIdx.x;
  if (e >= EP_EDGES) return;
  if (e < N_EDGES) { src[e] = (int)ei[e]; dst[e] = (int)ei[N_EDGES + e]; }
  else             { int v = e - N_EDGES; src[e] = v; dst[e] = v; }
}

// ---------------- dual GEMV via TDM + WMMA: xl = x@W1l+b1l, xr = x@W1r+b1r --
// One wave per 16-node tile. TDM stages the 16x256 f32 tile into LDS with a
// 4-DWORD row pad; WMMA A = 16x32 f16 slice, B columns {0: Wl, 1: Wr}.
__global__ void __launch_bounds__(32 * WAVES_PER_BLK)
k_gemv_wmma(const float* __restrict__ x,
            const float* __restrict__ Wl, const float* __restrict__ bl,
            const float* __restrict__ Wr, const float* __restrict__ br,
            float* __restrict__ xl, float* __restrict__ xr) {
  __shared__ float lds[WAVES_PER_BLK * 16 * ROW_PAD];
  const int wave = threadIdx.x >> 5;
  const int tile = blockIdx.x * WAVES_PER_BLK + wave;   // grid exact: no guard
  const int lane = threadIdx.x & 31;
  const int m  = lane & 15;                 // A row / B column / D column index
  const int hi = lane >> 4;                 // half-wave select
  float* slab = &lds[wave * 16 * ROW_PAD];
  const float* gsrc = x + (size_t)tile * 16 * F_IN;

#if defined(__gfx1250__) && __has_builtin(__builtin_amdgcn_tensor_load_to_lds)
  {
    // Tensor DMA descriptor (ISA 8.3/8.4): 2D tile 256 x 16 of 4B elements,
    // row stride 256, LDS pad 4 DWORDs every 256 DWORDs -> ROW_PAD=260 floats.
    unsigned lds_base = (unsigned)(size_t)(void*)slab;       // AS3 byte offset
    unsigned long long ga = (unsigned long long)(size_t)gsrc;
    v4u g0;
    g0[0] = 1u;                               // count=1, user descriptor
    g0[1] = lds_base;                         // lds_addr
    g0[2] = (unsigned)ga;                     // global_addr[31:0]
    g0[3] = (unsigned)(ga >> 32) | 0x80000000u; // global_addr[56:32] | type=2
    v8i g1;
    g1[0] = (int)((3u << 25) | (7u << 22) | (1u << 20) | (2u << 16));
            // pad_amount=4DW, pad_interval=256DW, pad_enable, data_size=4B
    g1[1] = (int)(256u << 16);                // tensor_dim0 = 256
    g1[2] = (int)(16u  << 16);                // tensor_dim1 = 16
    g1[3] = (int)(256u << 16);                // tile_dim0 = 256
    g1[4] = 16;                               // tile_dim1 = 16
    g1[5] = 256;                              // tensor_dim0_stride = 256
    g1[6] = 0;
    g1[7] = 0;
    v4i gz = {0, 0, 0, 0};
#if __clang_major__ >= 23
    v8i gz8 = {0, 0, 0, 0, 0, 0, 0, 0};
    __builtin_amdgcn_tensor_load_to_lds(g0, g1, gz, gz, gz8, 0);
#else
    __builtin_amdgcn_tensor_load_to_lds(g0, g1, gz, gz, 0);
#endif
    __builtin_amdgcn_s_wait_tensorcnt(0);
  }
#else
  // Fallback: cooperative coalesced copy (wave-local slab, block barrier).
  for (int i = lane; i < 16 * (F_IN / 4); i += 32) {
    int r = i >> 6;
    int c = (i & 63) << 2;
    float4 t = *(const float4*)(gsrc + r * F_IN + c);
    slab[r * ROW_PAD + c + 0] = t.x;
    slab[r * ROW_PAD + c + 1] = t.y;
    slab[r * ROW_PAD + c + 2] = t.z;
    slab[r * ROW_PAD + c + 3] = t.w;
  }
  __syncthreads();
#endif

  const float blv = bl[0], brv = br[0];
  const float* arow = &slab[m * ROW_PAD];
  v8f acc = {0.f, 0.f, 0.f, 0.f, 0.f, 0.f, 0.f, 0.f};

#pragma unroll
  for (int k0 = 0; k0 < F_IN; k0 += 32) {
    v16h a, b;
#pragma unroll
    for (int v = 0; v < 8; ++v) {
      // A (16x32, 16-bit): lanes 0-15 K=0..7 (v0..3) & 16..23 (v4..7); lanes 16-31 +8
      int ka = (v < 4) ? (hi * 8 + 2 * v) : (16 + hi * 8 + 2 * (v - 4));
      a[2 * v]     = (_Float16)arow[k0 + ka];
      a[2 * v + 1] = (_Float16)arow[k0 + ka + 1];
      // B (32x16, 16-bit): lane = column n, same K order as A
      int kb = (v < 4) ? (hi * 8 + 2 * v) : (16 + hi * 8 + 2 * (v - 4));
      float w0 = 0.f, w1 = 0.f;
      if (m == 0)      { w0 = Wl[k0 + kb]; w1 = Wl[k0 + kb + 1]; }
      else if (m == 1) { w0 = Wr[k0 + kb]; w1 = Wr[k0 + kb + 1]; }
      b[2 * v]     = (_Float16)w0;
      b[2 * v + 1] = (_Float16)w1;
    }
    acc = __builtin_amdgcn_wmma_f32_16x16x32_f16(false, a, false, b,
                                                 (short)0, acc, false, false);
  }
  // D (16x16 f32): lane = column n, VGPR v -> row v + hi*8
#pragma unroll
  for (int v = 0; v < 8; ++v) {
    int node = tile * 16 + v + hi * 8;
    if (m == 0)      xl[node] = acc[v] + blv;
    else if (m == 1) xr[node] = acc[v] + brv;
  }
}

// ---------------- layer-1 edge passes (score buffer lives in d_out's a1) ----
__global__ void k_score_max(const int* __restrict__ src, const int* __restrict__ dst,
                            const float* __restrict__ xl, const float* __restrict__ xr,
                            const float* __restrict__ att,
                            float* __restrict__ score, unsigned* __restrict__ smax) {
  int e = blockIdx.x * blockDim.x + threadIdx.x;
  if (e >= EP_EDGES) return;
  __builtin_prefetch(src + e + 4096, 0, 1);
  __builtin_prefetch(dst + e + 4096, 0, 1);
  int s = src[e], d = dst[e];
  float mv = xl[s] + xr[d];
  float ev = mv > 0.f ? mv : NEG_SLOPE * mv;
  float sc = ev * att[0];
  score[e] = sc;
  atomicMax(&smax[d], enc_f32(sc));
}

__global__ void k_ex_sum(const int* __restrict__ dst, const unsigned* __restrict__ smax,
                         float* __restrict__ score, float* __restrict__ denom) {
  int e = blockIdx.x * blockDim.x + threadIdx.x;
  if (e >= EP_EDGES) return;
  int d = dst[e];
  float ex = __expf(score[e] - dec_f32(smax[d]));
  score[e] = ex;
  atomicAdd(&denom[d], ex);
}

__global__ void k_alpha_agg(const int* __restrict__ src, const int* __restrict__ dst,
                            const float* __restrict__ xl, const float* __restrict__ denom,
                            float* __restrict__ score, float* __restrict__ h) {
  int e = blockIdx.x * blockDim.x + threadIdx.x;
  if (e >= EP_EDGES) return;
  int s = src[e], d = dst[e];
  float alpha = score[e] / denom[d];
  score[e] = alpha;                       // final a1 output
  atomicAdd(&h[d], alpha * xl[s]);
}

// ---------------- batchnorm statistics (sum, sumsq) -------------------------
__global__ void k_stats(const float* __restrict__ v, float* __restrict__ sum,
                        float* __restrict__ sumsq, int n) {
  float s = 0.f, q = 0.f;
  for (int i = blockIdx.x * blockDim.x + threadIdx.x; i < n;
       i += gridDim.x * blockDim.x) {
    float x = v[i]; s += x; q += x * x;
  }
  for (int o = 16; o; o >>= 1) { s += __shfl_down(s, o, 32); q += __shfl_down(q, o, 32); }
  if ((threadIdx.x & 31) == 0) { atomicAdd(sum, s); atomicAdd(sumsq, q); }
}

// ---------------- BN+ReLU, then layer-2 scalar transforms -------------------
__global__ void k_bn_l2(const float* __restrict__ h, const float* __restrict__ stats,
                        const float* __restrict__ gamma, const float* __restrict__ beta,
                        const float* __restrict__ W2l, const float* __restrict__ b2l,
                        const float* __restrict__ W2r, const float* __restrict__ b2r,
                        float* __restrict__ xl2, float* __restrict__ xr2) {
  int i = blockIdx.x * blockDim.x + threadIdx.x;
  if (i >= N_NODES) return;
  float mu  = stats[0] * (1.0f / N_NODES);
  float var = stats[1] * (1.0f / N_NODES) - mu * mu;
  float y = (h[i] - mu) * rsqrtf(var + BN_EPS) * gamma[0] + beta[0];
  y = y > 0.f ? y : 0.f;
  xl2[i] = y * W2l[0] + b2l[0];
  xr2[i] = y * W2r[0] + b2r[0];
}

// ---------------- layer-2 edge passes (recompute score; no edge buffer) -----
__device__ __forceinline__ float l2_score(const float* xl2, const float* xr2,
                                          float att, int s, int d) {
  float mv = xl2[s] + xr2[d];
  float ev = mv > 0.f ? mv : NEG_SLOPE * mv;
  return ev * att;
}

__global__ void k2_max(const int* __restrict__ src, const int* __restrict__ dst,
                       const float* __restrict__ xl2, const float* __restrict__ xr2,
                       const float* __restrict__ att, unsigned* __restrict__ smax) {
  int e = blockIdx.x * blockDim.x + threadIdx.x;
  if (e >= EP_EDGES) return;
  int s = src[e], d = dst[e];
  atomicMax(&smax[d], enc_f32(l2_score(xl2, xr2, att[0], s, d)));
}

__global__ void k2_sum(const int* __restrict__ src, const int* __restrict__ dst,
                       const float* __restrict__ xl2, const float* __restrict__ xr2,
                       const float* __restrict__ att, const unsigned* __restrict__ smax,
                       float* __restrict__ denom) {
  int e = blockIdx.x * blockDim.x + threadIdx.x;
  if (e >= EP_EDGES) return;
  int s = src[e], d = dst[e];
  float ex = __expf(l2_score(xl2, xr2, att[0], s, d) - dec_f32(smax[d]));
  atomicAdd(&denom[d], ex);
}

__global__ void k2_agg(const int* __restrict__ src, const int* __restrict__ dst,
                       const float* __restrict__ xl2, const float* __restrict__ xr2,
                       const float* __restrict__ att, const unsigned* __restrict__ smax,
                       const float* __restrict__ denom, float* __restrict__ h2) {
  int e = blockIdx.x * blockDim.x + threadIdx.x;
  if (e >= EP_EDGES) return;
  int s = src[e], d = dst[e];
  float ex = __expf(l2_score(xl2, xr2, att[0], s, d) - dec_f32(smax[d]));
  atomicAdd(&h2[d], (ex / denom[d]) * xl2[s]);
}

// ---------------- final BN+ReLU -> d_out[0:N] -------------------------------
__global__ void k_bn_out(const float* __restrict__ h, const float* __restrict__ stats,
                         const float* __restrict__ gamma, const float* __restrict__ beta,
                         float* __restrict__ out) {
  int i = blockIdx.x * blockDim.x + threadIdx.x;
  if (i >= N_NODES) return;
  float mu  = stats[0] * (1.0f / N_NODES);
  float var = stats[1] * (1.0f / N_NODES) - mu * mu;
  float y = (h[i] - mu) * rsqrtf(var + BN_EPS) * gamma[0] + beta[0];
  out[i] = y > 0.f ? y : 0.f;
}

extern "C" void kernel_launch(void* const* d_in, const int* in_sizes, int n_in,
                              void* d_out, int out_size, void* d_ws, size_t ws_size,
                              hipStream_t stream) {
  (void)in_sizes; (void)n_in; (void)out_size; (void)ws_size;
  const float*     x    = (const float*)d_in[0];
  const long long* ei   = (const long long*)d_in[1];
  const float* W1l  = (const float*)d_in[2];
  const float* b1l  = (const float*)d_in[3];
  const float* W1r  = (const float*)d_in[4];
  const float* b1r  = (const float*)d_in[5];
  const float* att1 = (const float*)d_in[6];
  // d_in[7] = bias1: shift-invariant under batchnorm -> no effect on outputs
  const float* gamma1 = (const float*)d_in[8];
  const float* beta1  = (const float*)d_in[9];
  const float* W2l  = (const float*)d_in[10];
  const float* b2l  = (const float*)d_in[11];
  const float* W2r  = (const float*)d_in[12];
  const float* b2r  = (const float*)d_in[13];
  const float* att2 = (const float*)d_in[14];
  // d_in[15] = bias2: also cancels under batchnorm
  const float* gamma2 = (const float*)d_in[16];
  const float* beta2  = (const float*)d_in[17];

  const size_t N = N_NODES;
  float*    ws    = (float*)d_ws;
  float*    xl    = ws;
  float*    xr    = ws + 1 * N;
  unsigned* smax1 = (unsigned*)(ws + 2 * N);
  float*    den1  = ws + 3 * N;
  float*    h1    = ws + 4 * N;
  float*    xl2   = ws + 5 * N;
  float*    xr2   = ws + 6 * N;
  unsigned* smax2 = (unsigned*)(ws + 7 * N);
  float*    den2  = ws + 8 * N;
  float*    h2    = ws + 9 * N;
  float*    stats = ws + 10 * N;                 // [sum1, sq1, sum2, sq2]
  int*      src   = (int*)(ws + 10 * N + 16);
  int*      dst   = src + EP_EDGES;

  float* out = (float*)d_out;
  float* a1  = out + N;   // layer-1 per-edge buffer: score -> ex -> alpha (output)

  const int TB = 256;
  const int gridN = (N_NODES + TB - 1) / TB;
  const int gridE = (EP_EDGES + TB - 1) / TB;
  const int gridT = TILES / WAVES_PER_BLK;       // exact: 12500 / 2 = 6250

  k_init     <<<gridN, TB, 0, stream>>>(smax1, den1, h1, smax2, den2, h2, stats);
  k_edges    <<<gridE, TB, 0, stream>>>(ei, src, dst);
  k_gemv_wmma<<<gridT, 32 * WAVES_PER_BLK, 0, stream>>>(x, W1l, b1l, W1r, b1r, xl, xr);

  k_score_max<<<gridE, TB, 0, stream>>>(src, dst, xl, xr, att1, a1, smax1);
  k_ex_sum   <<<gridE, TB, 0, stream>>>(dst, smax1, a1, den1);
  k_alpha_agg<<<gridE, TB, 0, stream>>>(src, dst, xl, den1, a1, h1);

  k_stats    <<<512,   TB, 0, stream>>>(h1, stats + 0, stats + 1, N_NODES);
  k_bn_l2    <<<gridN, TB, 0, stream>>>(h1, stats, gamma1, beta1,
                                        W2l, b2l, W2r, b2r, xl2, xr2);

  k2_max     <<<gridE, TB, 0, stream>>>(src, dst, xl2, xr2, att2, smax2);
  k2_sum     <<<gridE, TB, 0, stream>>>(src, dst, xl2, xr2, att2, smax2, den2);
  k2_agg     <<<gridE, TB, 0, stream>>>(src, dst, xl2, xr2, att2, smax2, den2, h2);

  k_stats    <<<512,   TB, 0, stream>>>(h2, stats + 2, stats + 3, N_NODES);
  k_bn_out   <<<gridN, TB, 0, stream>>>(h2, stats + 2, gamma2, beta2, out);
}
